// MultiHeadAttentionBlock_48421461295244
// MI455X (gfx1250) — compile-verified, matrix operand order checked by reading, not run on hardware
//
#include <hip/hip_runtime.h>
#include <hip/hip_bf16.h>

// ---------------------------------------------------------------------------
// MultiHeadAttentionBlock for MI455X (gfx1250), wave32 + WMMA bf16.
// B=8, C=64, L=256, D=64, H=4, HD=16.
// Pipeline: [conv3x3 QKV -> bf16 ws] -> [per-head attention] -> [1x1 conv +
// bias + residual + channel LayerNorm -> f32 out]
// ---------------------------------------------------------------------------

#define BDIM 8
#define CDIM 64
#define LDIM 256
#define DDIM 64

typedef __attribute__((ext_vector_type(16))) __bf16 v16bf;
typedef __attribute__((ext_vector_type(8)))  __bf16 v8bf;
typedef __attribute__((ext_vector_type(8)))  float  v8f;

static __device__ __forceinline__ v8f wmma_bf16(v16bf a, v16bf b, v8f c) {
  // (neg_a, A, neg_b, B, c_mod, C, reuse_a, reuse_b)
  return __builtin_amdgcn_wmma_f32_16x16x32_bf16(false, a, false, b, (short)0, c,
                                                 false, false);
}

// Load one 16x32 bf16 A/B fragment for this lane. `base` = row/col start + k0
// (K contiguous in memory). ISA 16-bit layout: lane half h holds K in
// {h*8+0..7, 16+h*8+0..7} -> two contiguous 16B loads.
static __device__ __forceinline__ v16bf load_frag(const __bf16* base, int khalf) {
  const __bf16* p = base + khalf * 8;
  v8bf lo = *(const v8bf*)(p);
  v8bf hi = *(const v8bf*)(p + 16);
  v16bf r;
#pragma unroll
  for (int i = 0; i < 8; ++i) { r[i] = lo[i]; r[8 + i] = hi[i]; }
  return r;
}

// ---------------------------------------------------------------------------
// Kernel 1: 3x3 SAME conv over (L,D), decomposed into 9 shifted GEMMs.
// grid = (L, B, 3 {q,k,v}), block = 128 (4 waves; wave w owns co-tile w).
// Dynamic LDS: weights [9][64co][64ci] bf16 + input rows [3][66d][64ci] bf16.
// Loop nest tap->kc->nt keeps live set tiny (no spills): each weight A-frag is
// read from LDS once and feeds 4 consecutive WMMAs on 4 resident accumulators.
// ---------------------------------------------------------------------------
__global__ void conv3x3_qkv(const float* __restrict__ q, const float* __restrict__ k,
                            const float* __restrict__ v, const float* __restrict__ Wq,
                            const float* __restrict__ Wk, const float* __restrict__ Wv,
                            __bf16* __restrict__ qb, __bf16* __restrict__ kb,
                            __bf16* __restrict__ vb) {
  extern __shared__ __bf16 smem[];
  __bf16* w_lds  = smem;              // 9*64*64
  __bf16* in_lds = smem + 9 * 4096;   // 3*66*64

  const int l = blockIdx.x, b = blockIdx.y, which = blockIdx.z;
  const float* in = (which == 0) ? q : (which == 1) ? k : v;
  const float* W  = (which == 0) ? Wq : (which == 1) ? Wk : Wv;
  __bf16* out     = (which == 0) ? qb : (which == 1) ? kb : vb;
  const int tid = threadIdx.x;

  // Stage weights transposed: W[co][ci][kh][kw] -> w_lds[tap][co][ci]
  for (int i = tid; i < 9 * 4096; i += 128) {
    int co = i / 576, rem = i % 576, ci = rem / 9, tap = rem % 9;
    w_lds[tap * 4096 + co * 64 + ci] = (__bf16)W[i];
  }
  // Stage input rows l-1..l+1, ci-contiguous, with zero-padded d columns.
  for (int row = 0; row < 3; ++row) {
    int lr = l + row - 1;
    __bf16* dst = in_lds + row * (66 * 64);
    if (lr < 0 || lr >= LDIM) {
      for (int i = tid; i < 66 * 64; i += 128) dst[i] = (__bf16)0.f;
    } else {
      const float* src = in + ((size_t)(b * CDIM) * LDIM + lr) * DDIM;
      for (int i = tid; i < 64 * 64; i += 128) {
        int ci = i >> 6, d = i & 63;
        dst[(d + 1) * 64 + ci] = (__bf16)src[(size_t)ci * LDIM * DDIM + d];
      }
      if (tid < 64) dst[tid] = (__bf16)0.f;                   // d = -1 pad
      else          dst[65 * 64 + (tid - 64)] = (__bf16)0.f;  // d = 64 pad
    }
  }
  __syncthreads();

  const int wave = tid >> 5, lane = tid & 31;
  const int nrow = lane & 15, khalf = lane >> 4;

  v8f acc[4] = {{}, {}, {}, {}};
  for (int tap = 0; tap < 9; ++tap) {
    const int kh = tap / 3, kw = tap % 3;
    const __bf16* arow  = w_lds + tap * 4096 + (wave * 16 + nrow) * 64;
    const __bf16* bbase = in_lds + kh * (66 * 64) + (nrow + kw) * 64;
#pragma unroll
    for (int kc = 0; kc < 2; ++kc) {
      v16bf af = load_frag(arow + kc * 32, khalf);  // loaded once, reused 4x
#pragma unroll
      for (int nt = 0; nt < 4; ++nt) {
        v16bf bf = load_frag(bbase + nt * (16 * 64) + kc * 32, khalf);
        acc[nt] = wmma_bf16(af, bf, acc[nt]);
      }
    }
  }
  // C layout: row co = wave*16 + khalf*8 + r, col d = nt*16 + nrow
#pragma unroll
  for (int nt = 0; nt < 4; ++nt) {
    __bf16* obase = out + (((size_t)b * CDIM) * LDIM + l) * DDIM + nt * 16 + nrow;
#pragma unroll
    for (int r = 0; r < 8; ++r) {
      int co = wave * 16 + khalf * 8 + r;
      obase[(size_t)co * LDIM * DDIM] = (__bf16)acc[nt][r];
    }
  }
}

// ---------------------------------------------------------------------------
// Kernel 2: per-(b, c=h*HD+hd) attention, one wave per 16 query rows.
// scores(16x256) in registers -> softmax -> LDS re-layout -> attn @ V.
// Output written channel-last: attn_t[b][l][d][c] (bf16).
// ---------------------------------------------------------------------------
__global__ void attention(const __bf16* __restrict__ qb, const __bf16* __restrict__ kb,
                          const __bf16* __restrict__ vb, __bf16* __restrict__ attn_t) {
  extern __shared__ __bf16 smem[];
  __bf16* a_lds = smem;             // [16][256] attn tile (row-major, K contiguous)
  __bf16* v_lds = smem + 16 * 256;  // [64 d][256 m] V transposed (K=m contiguous)

  const int lt = blockIdx.x, c = blockIdx.y, b = blockIdx.z;
  const size_t base = ((size_t)(b * CDIM + c)) * LDIM * DDIM;
  const __bf16* Q = qb + base;
  const __bf16* K = kb + base;
  const __bf16* V = vb + base;
  const int lane = threadIdx.x & 31;
  const int nrow = lane & 15, khalf = lane >> 4;

  // Stage V transposed into LDS.
  for (int m = lane; m < LDIM; m += 32)
    for (int d = 0; d < DDIM; ++d)
      v_lds[d * LDIM + m] = V[(size_t)m * DDIM + d];

  // Q fragments (A: rows l, K=d contiguous) straight from global.
  const __bf16* qrow = Q + (size_t)(lt * 16 + nrow) * DDIM;
  v16bf qf0 = load_frag(qrow, khalf);
  v16bf qf1 = load_frag(qrow + 32, khalf);

  v8f acc[16];
#pragma unroll
  for (int mt = 0; mt < 16; ++mt) {
    if (mt < 15) __builtin_prefetch(K + (size_t)((mt + 1) * 16 + nrow) * DDIM, 0, 0);
    const __bf16* krow = K + (size_t)(mt * 16 + nrow) * DDIM;  // B: col m, K=d contig
    v8f a = {};
    a = wmma_bf16(qf0, load_frag(krow, khalf), a);
    a = wmma_bf16(qf1, load_frag(krow + 32, khalf), a);
    acc[mt] = a;
  }

  // Softmax: row m = khalf*8 + r lives in vgpr r of each tile; columns span the
  // 16 lanes of this half across the 16 tiles. Reduce with xor masks 8..1.
  const float scale = 0.25f;  // 1/sqrt(HD)
#pragma unroll
  for (int r = 0; r < 8; ++r) {
    float mx = -3.0e38f;
#pragma unroll
    for (int mt = 0; mt < 16; ++mt) { float s = acc[mt][r] * scale; mx = s > mx ? s : mx; }
#pragma unroll
    for (int off = 8; off >= 1; off >>= 1) {
      float o = __shfl_xor(mx, off, 32); mx = o > mx ? o : mx;
    }
    float sum = 0.f;
#pragma unroll
    for (int mt = 0; mt < 16; ++mt) {
      float e = __expf(acc[mt][r] * scale - mx);
      acc[mt][r] = e; sum += e;
    }
#pragma unroll
    for (int off = 8; off >= 1; off >>= 1) sum += __shfl_xor(sum, off, 32);
    float inv = 1.f / sum;
    int lrow = khalf * 8 + r;
#pragma unroll
    for (int mt = 0; mt < 16; ++mt)
      a_lds[lrow * 256 + mt * 16 + nrow] = (__bf16)(acc[mt][r] * inv);
  }
  __syncthreads();

  // out = attn(16x256) @ V(256x64): kc outer so each attn A-frag is loaded
  // once and feeds the 4 resident d-tile accumulators.
  v8f oacc[4] = {{}, {}, {}, {}};
#pragma unroll
  for (int kc = 0; kc < 8; ++kc) {
    v16bf af = load_frag(a_lds + nrow * 256 + kc * 32, khalf);
#pragma unroll
    for (int dt = 0; dt < 4; ++dt) {
      v16bf bf = load_frag(v_lds + (dt * 16 + nrow) * LDIM + kc * 32, khalf);
      oacc[dt] = wmma_bf16(af, bf, oacc[dt]);
    }
  }
#pragma unroll
  for (int dt = 0; dt < 4; ++dt) {
#pragma unroll
    for (int r = 0; r < 8; ++r) {
      int lrow = lt * 16 + khalf * 8 + r;
      int d = dt * 16 + nrow;
      attn_t[(((size_t)b * LDIM + lrow) * DDIM + d) * CDIM + c] = (__bf16)oacc[dt][r];
    }
  }
}

// ---------------------------------------------------------------------------
// Kernel 3: 1x1 conv (64x64 GEMM) + bias + residual + LayerNorm over C, fused.
// grid = (L*D/16, B), one wave per 16-pixel tile.
// ---------------------------------------------------------------------------
__global__ void proj_ln(const __bf16* __restrict__ attn_t, const float* __restrict__ Wo,
                        const float* __restrict__ bo, const float* __restrict__ gamma,
                        const float* __restrict__ beta, const float* __restrict__ identity,
                        float* __restrict__ out) {
  __shared__ __bf16 w_lds[64 * 64];
  const int pt = blockIdx.x, b = blockIdx.y;
  const int lane = threadIdx.x & 31;
  const int nrow = lane & 15, khalf = lane >> 4;

  for (int i = lane; i < 4096; i += 32) w_lds[i] = (__bf16)Wo[i];
  __syncthreads();

  const int pix0 = pt * 16;
  // B fragments: attn_t channel-last -> K=ci contiguous per pixel column.
  const __bf16* brow = attn_t + ((size_t)b * LDIM * DDIM + pix0 + nrow) * CDIM;
  v16bf bf0 = load_frag(brow, khalf);
  v16bf bf1 = load_frag(brow + 32, khalf);

  v8f acc[4];
#pragma unroll
  for (int t = 0; t < 4; ++t) {
    const __bf16* arow = w_lds + (t * 16 + nrow) * 64;
    v8f a = {};
    a = wmma_bf16(load_frag(arow, khalf), bf0, a);
    a = wmma_bf16(load_frag(arow + 32, khalf), bf1, a);
    acc[t] = a;
  }

  // Epilogue: bias + residual, then LayerNorm over the 64 channels of each
  // pixel column. Column n's channels live in lanes n and n+16.
  const int pix = pix0 + nrow;
  float vals[32], s1 = 0.f, s2 = 0.f;
#pragma unroll
  for (int t = 0; t < 4; ++t)
#pragma unroll
    for (int r = 0; r < 8; ++r) {
      int co = t * 16 + khalf * 8 + r;
      float vv = acc[t][r] + bo[co] +
                 identity[((size_t)b * CDIM + co) * LDIM * DDIM + pix];
      vals[t * 8 + r] = vv; s1 += vv; s2 += vv * vv;
    }
  s1 += __shfl_xor(s1, 16, 32);
  s2 += __shfl_xor(s2, 16, 32);
  float mu  = s1 * (1.f / 64.f);
  float var = s2 * (1.f / 64.f) - mu * mu;
  float rs  = rsqrtf(var + 1e-5f);
#pragma unroll
  for (int t = 0; t < 4; ++t)
#pragma unroll
    for (int r = 0; r < 8; ++r) {
      int co = t * 16 + khalf * 8 + r;
      out[((size_t)b * CDIM + co) * LDIM * DDIM + pix] =
          (vals[t * 8 + r] - mu) * rs * gamma[co] + beta[co];
    }
}

// ---------------------------------------------------------------------------
extern "C" void kernel_launch(void* const* d_in, const int* in_sizes, int n_in,
                              void* d_out, int out_size, void* d_ws, size_t ws_size,
                              hipStream_t stream) {
  (void)in_sizes; (void)n_in; (void)out_size; (void)ws_size;
  const float* q     = (const float*)d_in[0];
  const float* k     = (const float*)d_in[1];
  const float* v     = (const float*)d_in[2];
  const float* Wq    = (const float*)d_in[3];
  const float* Wk    = (const float*)d_in[4];
  const float* Wv    = (const float*)d_in[5];
  const float* Wo    = (const float*)d_in[6];
  const float* bo    = (const float*)d_in[7];
  const float* gamma = (const float*)d_in[8];
  const float* beta  = (const float*)d_in[9];

  const size_t NE = (size_t)BDIM * CDIM * LDIM * DDIM;  // 8,388,608 elements
  char* ws = (char*)d_ws;
  __bf16* qb     = (__bf16*)(ws);
  __bf16* kb     = (__bf16*)(ws + NE * 2);
  __bf16* vb     = (__bf16*)(ws + NE * 4);
  __bf16* attn_t = (__bf16*)(ws + NE * 6);

  const size_t sh_conv = (size_t)(9 * 4096 + 3 * 66 * 64) * sizeof(__bf16);  // ~97 KB
  conv3x3_qkv<<<dim3(LDIM, BDIM, 3), 128, sh_conv, stream>>>(q, k, v, Wq, Wk, Wv,
                                                             qb, kb, vb);

  const size_t sh_attn = (size_t)(16 * 256 + 64 * 256) * sizeof(__bf16);  // 40 KB
  attention<<<dim3(LDIM / 16, CDIM, BDIM), 32, sh_attn, stream>>>(qb, kb, vb, attn_t);

  proj_ln<<<dim3(LDIM * DDIM / 16, BDIM), 32, 0, stream>>>(attn_t, Wo, bo, gamma, beta,
                                                           q, (float*)d_out);
}
